// SoftQuantize_60103772340316
// MI455X (gfx1250) — matrix operand order repaired and checked by reading, not compiled, not run on hardware
//
#include <hip/hip_runtime.h>
#include <hip/hip_bf16.h>
#include <math.h>

typedef __attribute__((ext_vector_type(16))) _Float16 v16h;
typedef __attribute__((ext_vector_type(8)))  float    v8f;

#define Bn   32
#define Cn   64
#define Hn   64
#define Wn   64
#define Mn   512
#define Dn   64
#define Kn   (Bn*Hn*Wn)          /* 131072 */
#define HWn  (Hn*Wn)             /* 4096   */
#define CHWn (Cn*HWn)            /* 262144 */

/* workspace layout (bytes). ebuf and esq are contiguous so one async LDS
   staging pass covers both (64KB + 2KB). */
#define WS_IDX   0
#define WS_EBUF  (Kn*4)                    /* 524288: packed f16 B tiles (64KB) */
#define WS_ESQ   (WS_EBUF + 64*32*32)      /* 589824: ||e||^2 per code (2KB)    */
#define WS_HIST  (WS_ESQ + Mn*4)           /* 591872: one-hot histogram         */
#define WS_KL    (WS_HIST + Mn*4)          /* 593920: KL accumulator            */

#define STAGE_BYTES (64*32*32 + Mn*4)      /* 67584 */

/* output layout (floats): q, KL, embed_ind, perplexity */
#define OUT_KL   (Bn*Cn*Hn*Wn)             /* 8388608 */
#define OUT_IND  (OUT_KL + 1)
#define OUT_PERP (OUT_IND + Kn)

#define LOG_M 6.238324625039508f           /* ln(512) */

/* ---- pack embedding into CDNA5 16-bit B-matrix layout (32x16 per K-chunk) ----
   B order used here (16-bit, 32x16; contiguous K, same as A): lanes 0-15 hold column N=lane, K=0..15 in halves;
   lanes 16-31 hold column N=lane-16, K=16..31.                                 */
__global__ void vq_pack_emb(const float* __restrict__ emb,
                            _Float16* __restrict__ ebuf) {
  int wid   = blockIdx.x;          /* 0..63 : 32 m-tiles x 2 K-chunks */
  int mtile = wid >> 1;
  int chunk = wid & 1;
  int lane  = threadIdx.x;         /* 0..31 */
  int m     = mtile * 16 + (lane & 15);
  v16h v;
#pragma unroll
  for (int h = 0; h < 16; ++h) {
    int K = (lane < 16) ? h : (16 + h);
    v[h] = (_Float16)emb[m * Dn + chunk * 32 + K];
  }
  *(v16h*)(ebuf + ((mtile * 2 + chunk) * 32 + lane) * 16) = v;
}

/* ---- ||e_m||^2 + zero the atomic accumulators (ws is poisoned 0xAA) ---- */
__global__ void vq_esq_init(const float* __restrict__ emb,
                            float* __restrict__ esq,
                            unsigned* __restrict__ hist,
                            float* __restrict__ klacc) {
  int m = blockIdx.x * blockDim.x + threadIdx.x;
  if (m < Mn) {
    float s = 0.f;
#pragma unroll 8
    for (int d = 0; d < Dn; ++d) { float e = emb[m * Dn + d]; s += e * e; }
    esq[m]  = s;
    hist[m] = 0u;
  }
  if (m == 0) *klacc = 0.f;
}

/* ---- main: one wave per 16-row tile; async LDS-staged B; fused WMMA +
        online softmax/argmax over M=512 codes ---- */
__global__ void __launch_bounds__(128)
vq_main(const float* __restrict__ x,
        const char* __restrict__ estage,   /* ebuf (64KB) + esq (2KB) */
        int* __restrict__ idxbuf,
        unsigned* __restrict__ hist,
        float* __restrict__ klacc,
        float* __restrict__ out_ind) {
  extern __shared__ __align__(16) char smem[];
  const int lane    = threadIdx.x & 31;
  const int rowtile = blockIdx.x * 4 + (threadIdx.x >> 5);   /* 0..8191 */

  /* ---- stage packed B + ||e||^2 into LDS via CDNA5 async copy ----
     4224 lane-transfers of 16B = 66KB; 33 wave-wide async ops per wave. */
  {
    uint32_t lbase = (uint32_t)(uintptr_t)&smem[0];
#pragma unroll 4
    for (int i = threadIdx.x; i < STAGE_BYTES / 16; i += 128) {
      uint32_t    ldst = lbase + (uint32_t)i * 16u;
      const char* g    = estage + (size_t)i * 16;
      asm volatile("global_load_async_to_lds_b128 %0, %1, off"
                   :: "v"(ldst), "v"(g) : "memory");
    }
    asm volatile("s_wait_asynccnt 0" ::: "memory");
  }

  /* Build A (16 rows x 64 K) in the 16-bit A order used here (contiguous K, as B) after the
     async copy completes. Row M = lane&15; lane<16 holds K 0..15 of
     each 32-chunk, lane>=16 holds K 16..31. x stride HWn per d. */
  const int    krow = rowtile * 16 + (lane & 15);
  const float* xb   = x + (krow >> 12) * CHWn + (krow & (HWn - 1));
  v16h a0, a1;
#pragma unroll
  for (int h = 0; h < 16; ++h) {
    int K = (lane < 16) ? h : (16 + h);
    a0[h] = (_Float16)xb[K * HWn];
    a1[h] = (_Float16)xb[(K + 32) * HWn];
  }

  __syncthreads();   /* LDS staging visible to all 4 waves */

  const _Float16* bsm = (const _Float16*)smem;
  const float*    esm = (const float*)(smem + 64 * 32 * 32);

  /* online-softmax state per C-fragment row (8 rows per lane) */
  float bs[8], S0[8], S1[8]; int bi[8];
#pragma unroll
  for (int r = 0; r < 8; ++r) { bs[r] = -__builtin_inff(); S0[r] = 0.f; S1[r] = 0.f; bi[r] = 0; }

  for (int t = 0; t < 32; ++t) {
    v16h b0 = *(const v16h*)(bsm + ((t * 2) * 32 + lane) * 16);
    v16h b1 = *(const v16h*)(bsm + ((t * 2 + 1) * 32 + lane) * 16);
    float es = esm[t * 16 + (lane & 15)];

    v8f c = {};
    c = __builtin_amdgcn_wmma_f32_16x16x32_f16(false, a0, false, b0,
                                               (short)0, c, false, false);
    c = __builtin_amdgcn_wmma_f32_16x16x32_f16(false, a1, false, b1,
                                               (short)0, c, false, false);

    const int col = t * 16 + (lane & 15);
#pragma unroll
    for (int r = 0; r < 8; ++r) {
      /* score = 2 x.e - ||e||^2  (||x||^2 cancels in argmin & softmax) */
      float s  = 2.f * c[r] - es;
      float nm = fmaxf(bs[r], s);
      float ea = __expf(bs[r] - nm);
      float eb = __expf(s - nm);
      S0[r] = S0[r] * ea + eb;
      S1[r] = S1[r] * ea + eb * s;
      bi[r] = (s > bs[r]) ? col : bi[r];
      bs[r] = nm;
    }
  }

  /* reduce across the 16 lanes of each half (C layout: rows r / r+8) */
#pragma unroll
  for (int r = 0; r < 8; ++r) {
#pragma unroll
    for (int m = 1; m < 16; m <<= 1) {
      float obs = __shfl_xor(bs[r], m, 32);
      float oS0 = __shfl_xor(S0[r], m, 32);
      float oS1 = __shfl_xor(S1[r], m, 32);
      int   obi = __shfl_xor(bi[r], m, 32);
      float nm  = fmaxf(bs[r], obs);
      float ea  = __expf(bs[r] - nm);
      float eb  = __expf(obs - nm);
      S0[r] = S0[r] * ea + oS0 * eb;
      S1[r] = S1[r] * ea + oS1 * eb;
      bi[r] = (obs > bs[r]) ? obi : bi[r];
      bs[r] = nm;
    }
  }

  /* one lane per row emits index / histogram / KL term */
  float klloc = 0.f;
#pragma unroll
  for (int r = 0; r < 8; ++r) {
    if ((lane & 15) == r) {
      int row = rowtile * 16 + ((lane < 16) ? r : r + 8);
      /* KL_row = E[s] - logsumexp + log M  (mean taken in finalize) */
      klloc += (S1[r] / S0[r]) - (bs[r] + __logf(S0[r])) + LOG_M;
      atomicAdd(&hist[bi[r]], 1u);
      idxbuf[row]  = bi[r];
      out_ind[row] = (float)bi[r];
    }
  }
#pragma unroll
  for (int m = 1; m < 32; m <<= 1) klloc += __shfl_xor(klloc, m, 32);
  if (lane == 0) atomicAdd(klacc, klloc);
}

/* ---- coalesced codebook gather: q[b,d,h,w] = emb[idx[k], d] ---- */
__global__ void vq_gather(const float* __restrict__ emb,
                          const int* __restrict__ idxbuf,
                          float* __restrict__ q) {
  int t = blockIdx.x * blockDim.x + threadIdx.x;   /* 0..8388607 */
  int k = t & (Kn - 1);
  int d = t >> 17;
  q[(k >> 12) * CHWn + d * HWn + (k & (HWn - 1))] = emb[idxbuf[k] * Dn + d];
}

/* ---- scalars: KL mean, perplexity from histogram ---- */
__global__ void vq_finalize(const unsigned* __restrict__ hist,
                            const float* __restrict__ klacc,
                            float* __restrict__ out) {
  __shared__ float sh[512];
  int m = threadIdx.x;
  float avg = (float)hist[m] * (1.f / (float)Kn);
  sh[m] = avg * __logf(avg + 1e-10f);
  __syncthreads();
  for (int s = 256; s > 0; s >>= 1) {
    if (m < s) sh[m] += sh[m + s];
    __syncthreads();
  }
  if (m == 0) {
    out[OUT_PERP] = __expf(-sh[0]);
    out[OUT_KL]   = (*klacc) * (1.f / ((float)Kn * (float)Mn));
  }
}

extern "C" void kernel_launch(void* const* d_in, const int* in_sizes, int n_in,
                              void* d_out, int out_size, void* d_ws, size_t ws_size,
                              hipStream_t stream) {
  const float* x   = (const float*)d_in[0];
  const float* emb = (const float*)d_in[1];
  float*       out = (float*)d_out;
  char*        ws  = (char*)d_ws;

  int*       idxbuf = (int*)(ws + WS_IDX);
  _Float16*  ebuf   = (_Float16*)(ws + WS_EBUF);
  float*     esq    = (float*)(ws + WS_ESQ);
  unsigned*  hist   = (unsigned*)(ws + WS_HIST);
  float*     klacc  = (float*)(ws + WS_KL);

  vq_pack_emb<<<64, 32, 0, stream>>>(emb, ebuf);
  vq_esq_init<<<2, 256, 0, stream>>>(emb, esq, hist, klacc);
  vq_main<<<Kn / 16 / 4, 128, STAGE_BYTES, stream>>>(x, ws + WS_EBUF, idxbuf,
                                                     hist, klacc,
                                                     out + OUT_IND);
  vq_gather<<<(Bn * Cn * Hn * Wn) / 256, 256, 0, stream>>>(emb, idxbuf,
                                                           out /* q */);
  vq_finalize<<<1, 512, 0, stream>>>(hist, klacc, out);
}
